// SOTAAttentionRouter_63196148793905
// MI455X (gfx1250) — compile-verified, matrix operand order checked by reading, not run on hardware
//
#include <hip/hip_runtime.h>
#include <hip/hip_bf16.h>
#include <math.h>

// ---------------------------------------------------------------------------
// Problem constants (from reference): B=2, S=2048, H=2048, NH=16, NKV=4, HD=128
// ---------------------------------------------------------------------------
#define CB   2
#define CS   2048
#define CH   2048
#define CNH  16
#define CNKV 4
#define CHD  128
#define CM   (CB * CS)            // 4096 rows
#define CQN  (CNH * CHD)          // 2048
#define CKVN (CNKV * CHD)         // 512

typedef unsigned short     u16;
typedef unsigned int       u32;
typedef unsigned long long u64;
typedef __attribute__((ext_vector_type(16))) __bf16 v16bf;
typedef __attribute__((ext_vector_type(8)))  float  v8f;
typedef __attribute__((ext_vector_type(8)))  u32    v8u;
typedef __attribute__((ext_vector_type(4)))  u32    u32x4;
typedef int i32x4g __attribute__((vector_size(16)));   // matches async-LDS builtin param

// Feature probes (toolchain-dependent builtins)
#if defined(__has_builtin)
#  if __has_builtin(__builtin_amdgcn_global_load_async_to_lds_b128)
#    define HAVE_ASYNC_LDS 1
#  endif
#  if __has_builtin(__builtin_amdgcn_s_wait_asynccnt)
#    define HAVE_WAIT_ASYNC 1
#  endif
#endif

__device__ __forceinline__ u16 f2bf(float f) {
    u32 u = __builtin_bit_cast(u32, f);
    u += 0x7FFFu + ((u >> 16) & 1u);          // round-to-nearest-even
    return (u16)(u >> 16);
}
__device__ __forceinline__ float bf2f(u16 h) {
    u32 u = ((u32)h) << 16;
    return __builtin_bit_cast(float, u);
}

// D = A*B + C   (16x16x32 bf16, fp32 accum) — wave32 only
__device__ __forceinline__ v8f wmma_bf16(v8u a, v8u b, v8f c) {
    return __builtin_amdgcn_wmma_f32_16x16x32_bf16(
        false, __builtin_bit_cast(v16bf, a),
        false, __builtin_bit_cast(v16bf, b),
        (short)0, c, false, false);
}

// Load an 8-dword A/B fragment where K-pairs are contiguous u16:
// dwords [0..3] = K(kbase..kbase+7), dwords [4..7] = K(kbase+16..kbase+23).
__device__ __forceinline__ v8u load_frag_pair(const u16* p) {
    u32x4 lo = *(const u32x4*)(p);
    u32x4 hi = *(const u32x4*)(p + 16);
    v8u r;
    r[0] = lo.x; r[1] = lo.y; r[2] = lo.z; r[3] = lo.w;
    r[4] = hi.x; r[5] = hi.y; r[6] = hi.z; r[7] = hi.w;
    return r;
}

// Generic -> LDS(as3) offset for TDM descriptors
__device__ __forceinline__ u32 lds_addr_of(void* p) {
    return (u32)(uintptr_t)(__attribute__((address_space(3))) void*)p;
}

// ---------------------------------------------------------------------------
// Tensor Data Mover: 2-D tile load Global -> LDS (CDNA5 ISA)
// D# group0 (4 SGPRs): count=1 | lds_addr | global_addr[56:0] | type=2
// D# group1 (8 SGPRs): data_size=2B, tensor_dim0/1, tile_dim0/1, dim0_stride
// Tracked by TENSORcnt; wait via s_wait_tensorcnt.
// ---------------------------------------------------------------------------
__device__ __forceinline__ void tdm_load_2d_to_lds(u32 lds_off, const u16* gsrc,
                                                   u32 tile_d0, u32 tile_d1,
                                                   u32 stride_elems) {
    u64 ga = (u64)(uintptr_t)gsrc;
    u32x4 g0;
    g0[0] = (u32)__builtin_amdgcn_readfirstlane(1u);                         // count=1
    g0[1] = (u32)__builtin_amdgcn_readfirstlane(lds_off);                    // lds_addr
    g0[2] = (u32)__builtin_amdgcn_readfirstlane((u32)ga);                    // gaddr[31:0]
    g0[3] = (u32)__builtin_amdgcn_readfirstlane((u32)(ga >> 32) | (2u << 30)); // gaddr hi | type=2

    const u32 td0 = stride_elems;        // tensor row length (elements)
    const u32 td1 = 0x40000000u;         // effectively unbounded dim1
    v8u g1;
    g1[0] = 1u << 16;                                  // workgroup_mask=0, data_size=1 (2B)
    g1[1] = (td0 & 0xFFFFu) << 16;                     // tensor_dim0[15:0]
    g1[2] = (td0 >> 16) | ((td1 & 0xFFFFu) << 16);     // dim0[31:16] | dim1[15:0]
    g1[3] = (td1 >> 16) | (tile_d0 << 16);             // dim1[31:16] | tile_dim0
    g1[4] = tile_d1;                                   // tile_dim1 (tile_dim2=0)
    g1[5] = stride_elems;                              // tensor_dim0_stride[31:0]
    g1[6] = 0;                                         // stride[47:32] | dim1_stride lo
    g1[7] = 0;
#pragma unroll
    for (int i = 0; i < 8; ++i) g1[i] = (u32)__builtin_amdgcn_readfirstlane(g1[i]);

    asm volatile("tensor_load_to_lds %0, %1" :: "s"(g0), "s"(g1) : "memory");
}

__device__ __forceinline__ void wait_tensorcnt0() {
    __builtin_amdgcn_s_wait_tensorcnt(0);
}

// ---------------------------------------------------------------------------
// 1) fp32 -> bf16 conversion
// ---------------------------------------------------------------------------
__global__ void cvt_f32_bf16_kernel(const float* __restrict__ src,
                                    u16* __restrict__ dst, int n) {
    int i = blockIdx.x * blockDim.x + threadIdx.x;
    if (i < n) dst[i] = f2bf(src[i]);
}

// ---------------------------------------------------------------------------
// 2) Tiled BF16 WMMA GEMM: C[M,N] = A[M,K] @ B[K,N]
//    block = 256 threads (8 waves), tile 64x64, K-step 32.
//    A tile copied global->LDS via GLOBAL_LOAD_ASYNC_TO_LDS_B128 when available.
// ---------------------------------------------------------------------------
template <bool OUT_F32>
__global__ __launch_bounds__(256) void gemm_bf16_wmma_kernel(
    const u16* __restrict__ A, const u16* __restrict__ B,
    float* __restrict__ Cf, u16* __restrict__ Cb,
    int M, int N, int K) {
    __shared__ alignas(16) u16 ldsA[64 * 32];
    __shared__ alignas(16) u16 ldsB[64 * 32];

    const int t    = threadIdx.x;
    const int wave = t >> 5;
    const int lane = t & 31;
    const int lrow = lane & 15;
    const int lhi  = lane >> 4;

    const int mblk = blockIdx.x;
    const int nblk = blockIdx.y;
    const int mt   = (wave & 3) * 16;
    const int nb0  = (wave >> 2) * 32;

    v8f c0 = {}; v8f c1 = {};

    const int arow = t >> 2, aq = t & 3;          // A tile: 256 x 16B
    const int bkr  = t >> 3, bnc = (t & 7) * 8;   // B tile: 256 x 16B

    for (int kb = 0; kb < K; kb += 32) {
        // --- A tile: direct global->LDS async copy (16B per lane) ---
        {
            const u16* gp = A + (size_t)(mblk * 64 + arow) * K + kb + aq * 8;
#if defined(HAVE_ASYNC_LDS)
            __builtin_amdgcn_global_load_async_to_lds_b128(
                (__attribute__((address_space(1))) i32x4g*)(void*)(u16*)gp,
                (__attribute__((address_space(3))) i32x4g*)(void*)(ldsA + t * 8),
                0, 0);
#else
            ((u32x4*)ldsA)[t] = *(const u32x4*)gp;
#endif
            if (kb + 32 < K)
                __builtin_prefetch(A + (size_t)(mblk * 64 + arow) * K + kb + 32 + aq * 8, 0, 0);
        }
        // --- B tile: load + transpose into n-major (fragment-ready) layout ---
        {
            const u32x4* gp = (const u32x4*)(B + (size_t)(kb + bkr) * N + nblk * 64 + bnc);
            u32x4 v = *gp;
            u16 e[8];
            e[0] = (u16)v.x; e[1] = (u16)(v.x >> 16);
            e[2] = (u16)v.y; e[3] = (u16)(v.y >> 16);
            e[4] = (u16)v.z; e[5] = (u16)(v.z >> 16);
            e[6] = (u16)v.w; e[7] = (u16)(v.w >> 16);
#pragma unroll
            for (int j = 0; j < 8; ++j) ldsB[(bnc + j) * 32 + bkr] = e[j];
            if (kb + 32 < K)
                __builtin_prefetch(B + (size_t)(kb + 32 + bkr) * N + nblk * 64 + bnc, 0, 0);
        }
#if defined(HAVE_ASYNC_LDS)
#  if defined(HAVE_WAIT_ASYNC)
        __builtin_amdgcn_s_wait_asynccnt(0);
#  else
        asm volatile("s_wait_asynccnt 0x0" ::: "memory");
#  endif
#endif
        __syncthreads();

        v8u a  = load_frag_pair(ldsA + (mt + lrow) * 32 + lhi * 8);
        v8u b0 = load_frag_pair(ldsB + (nb0 + lrow) * 32 + lhi * 8);
        v8u b1 = load_frag_pair(ldsB + (nb0 + 16 + lrow) * 32 + lhi * 8);
        c0 = wmma_bf16(a, b0, c0);
        c1 = wmma_bf16(a, b1, c1);
        __syncthreads();
    }

#pragma unroll
    for (int i = 0; i < 8; ++i) {
        size_t row  = (size_t)(mblk * 64 + mt + i + 8 * lhi);
        size_t col0 = (size_t)(nblk * 64 + nb0 + lrow);
        if (OUT_F32) {
            Cf[row * N + col0]      = c0[i];
            Cf[row * N + col0 + 16] = c1[i];
        } else {
            Cb[row * N + col0]      = f2bf(c0[i]);
            Cb[row * N + col0 + 16] = f2bf(c1[i]);
        }
    }
}

// ---------------------------------------------------------------------------
// 3) RoPE in-place on bf16 [rows, nh, 128]; pos = row % S
// ---------------------------------------------------------------------------
__global__ void rope_bf16_kernel(u16* __restrict__ x, int rows, int nh) {
    int i = blockIdx.x * blockDim.x + threadIdx.x;
    int total = rows * nh * 64;
    if (i >= total) return;
    int d = i & 63;
    int tmp = i >> 6;
    int h = tmp % nh;
    int row = tmp / nh;
    int pos = row & (CS - 1);
    float inv = __expf(-(float)d * (9.210340371976184f / 64.0f));
    float ang = (float)pos * inv;
    float c = __cosf(ang), s = __sinf(ang);
    size_t base = (size_t)row * nh * CHD + (size_t)h * CHD + d;
    float x1 = bf2f(x[base]);
    float x2 = bf2f(x[base + 64]);
    x[base]      = f2bf(x1 * c - x2 * s);
    x[base + 64] = f2bf(x2 * c + x1 * s);
}

// ---------------------------------------------------------------------------
// 4) Flash attention, causal, GQA. One wave per 16 query rows, 2 waves/block.
//    K/V 32x128 tiles staged into LDS by the Tensor Data Mover each KV step.
// ---------------------------------------------------------------------------
__global__ __launch_bounds__(64) void flash_attn_wmma_kernel(
    const u16* __restrict__ Q, const u16* __restrict__ Kb,
    const u16* __restrict__ Vb, u16* __restrict__ O) {
    __shared__ alignas(16) u16 ldsK[2][32 * 128];
    __shared__ alignas(16) u16 ldsV[2][32 * 128];
    __shared__ alignas(16) u16 ldsP[2][16 * 32];

    const int lane  = threadIdx.x & 31;
    const int wv    = threadIdx.x >> 5;
    const int wg    = blockIdx.x * 2 + wv;
    const int qtile = wg & ((CS / 16) - 1);        // 0..127
    const int bh    = wg >> 7;
    const int b     = bh >> 4;
    const int h     = bh & 15;
    const int kvh   = h >> 2;                      // GQA: 4 q-heads per kv-head
    const int lrow  = lane & 15;
    const int lhi   = lane >> 4;

    const size_t qrowbase = (size_t)(b * CS + qtile * 16);
    const u32 ldsK_off = lds_addr_of(&ldsK[wv][0]);
    const u32 ldsV_off = lds_addr_of(&ldsV[wv][0]);

    // Q fragments: 4 chunks of K=32 across HD=128
    v8u qf[4];
    {
        const u16* qp = Q + (qrowbase + lrow) * CQN + (size_t)h * CHD + lhi * 8;
#pragma unroll
        for (int kc = 0; kc < 4; ++kc) qf[kc] = load_frag_pair(qp + kc * 32);
    }

    v8f o[8];
#pragma unroll
    for (int dc = 0; dc < 8; ++dc) { v8f z = {}; o[dc] = z; }
    float run_m[8], run_l[8];
#pragma unroll
    for (int i = 0; i < 8; ++i) { run_m[i] = -1e30f; run_l[i] = 0.0f; }

    const float scaling = 0.08838834764831845f;    // 1/sqrt(128)
    const int numkt = (qtile * 16 + 16 + 31) >> 5; // causal KV tile count

    for (int kt = 0; kt < numkt; ++kt) {
        const int key0 = kt * 32;

        // ----- TDM: stage K and V 32x128 tiles into this wave's LDS -----
        const u16* kg = Kb + (size_t)(b * CS + key0) * CKVN + (size_t)kvh * CHD;
        const u16* vg = Vb + (size_t)(b * CS + key0) * CKVN + (size_t)kvh * CHD;
        asm volatile("s_wait_dscnt 0x0" ::: "memory");   // tile buffers reusable
        tdm_load_2d_to_lds(ldsK_off, kg, CHD, 32, CKVN);
        tdm_load_2d_to_lds(ldsV_off, vg, CHD, 32, CKVN);
        wait_tensorcnt0();

        // ----- S tiles: Q @ K^T for 2 sub-tiles of 16 keys (from LDS) -----
        v8f s0 = {}; v8f s1 = {};
        {
            const u16* kp0 = &ldsK[wv][0] + (size_t)lrow * CHD + lhi * 8;
            const u16* kp1 = &ldsK[wv][0] + (size_t)(16 + lrow) * CHD + lhi * 8;
#pragma unroll
            for (int kc = 0; kc < 4; ++kc) {
                v8u bk0 = load_frag_pair(kp0 + kc * 32);
                v8u bk1 = load_frag_pair(kp1 + kc * 32);
                s0 = wmma_bf16(qf[kc], bk0, s0);
                s1 = wmma_bf16(qf[kc], bk1, s1);
            }
        }

        // ----- scale + causal mask + row-max (reduce over 16 lanes) -----
        float tm[8];
#pragma unroll
        for (int i = 0; i < 8; ++i) {
            int rg  = qtile * 16 + i + 8 * lhi;
            int k0g = key0 + lrow;
            int k1g = key0 + 16 + lrow;
            float v0 = (k0g <= rg) ? s0[i] * scaling : -1e30f;
            float v1 = (k1g <= rg) ? s1[i] * scaling : -1e30f;
            s0[i] = v0; s1[i] = v1;
            float mx = fmaxf(v0, v1);
            mx = fmaxf(mx, __shfl_xor(mx, 1, 32));
            mx = fmaxf(mx, __shfl_xor(mx, 2, 32));
            mx = fmaxf(mx, __shfl_xor(mx, 4, 32));
            mx = fmaxf(mx, __shfl_xor(mx, 8, 32));
            tm[i] = mx;
        }

        // ----- online softmax update -----
        float p0[8], p1[8], corr[8];
#pragma unroll
        for (int i = 0; i < 8; ++i) {
            float nm = fmaxf(run_m[i], tm[i]);
            corr[i] = __expf(run_m[i] - nm);
            float e0 = __expf(s0[i] - nm);
            float e1 = __expf(s1[i] - nm);
            float rs = e0 + e1;
            rs += __shfl_xor(rs, 1, 32);
            rs += __shfl_xor(rs, 2, 32);
            rs += __shfl_xor(rs, 4, 32);
            rs += __shfl_xor(rs, 8, 32);
            run_l[i] = run_l[i] * corr[i] + rs;
            run_m[i] = nm;
            p0[i] = e0; p1[i] = e1;
        }
#pragma unroll
        for (int dc = 0; dc < 8; ++dc)
#pragma unroll
            for (int i = 0; i < 8; ++i) o[dc][i] *= corr[i];

        // ----- P -> LDS (row-major 16x32 bf16), reread as A-fragment -----
        u16* lp = ldsP[wv];
#pragma unroll
        for (int i = 0; i < 8; ++i) {
            int m = i + 8 * lhi;
            lp[m * 32 + lrow]      = f2bf(p0[i]);
            lp[m * 32 + 16 + lrow] = f2bf(p1[i]);
        }
        // DS ops from the same wave execute in order -> no barrier needed.
        v8u pf = load_frag_pair(lp + lrow * 32 + lhi * 8);

        // ----- P @ V: 8 output d-chunks of 16 (V gathered from LDS) -----
        const u16* vp = &ldsV[wv][0];
#pragma unroll
        for (int dc = 0; dc < 8; ++dc) {
            v8u bvf;
#pragma unroll
            for (int j = 0; j < 4; ++j) {
                int k = lhi * 8 + 2 * j;
                u32 w0 = (u32)vp[k * CHD + dc * 16 + lrow] |
                         ((u32)vp[(k + 1) * CHD + dc * 16 + lrow] << 16);
                u32 w1 = (u32)vp[(k + 16) * CHD + dc * 16 + lrow] |
                         ((u32)vp[(k + 17) * CHD + dc * 16 + lrow] << 16);
                bvf[j]     = w0;
                bvf[4 + j] = w1;
            }
            o[dc] = wmma_bf16(pf, bvf, o[dc]);
        }
    }

    // ----- normalize + store attn output (bf16, [B*S, NH*HD]) -----
#pragma unroll
    for (int dc = 0; dc < 8; ++dc)
#pragma unroll
        for (int i = 0; i < 8; ++i) {
            int m = i + 8 * lhi;
            float val = o[dc][i] / run_l[i];
            O[(qrowbase + m) * CQN + (size_t)h * CHD + dc * 16 + lrow] = f2bf(val);
        }
}

// ---------------------------------------------------------------------------
// Launch
// ---------------------------------------------------------------------------
extern "C" void kernel_launch(void* const* d_in, const int* in_sizes, int n_in,
                              void* d_out, int out_size, void* d_ws, size_t ws_size,
                              hipStream_t stream) {
    const float* hs = (const float*)d_in[0];
    // d_in[1] = position_ids (arange; RoPE uses row % S directly)
    const float* Wq = (const float*)d_in[2];
    const float* Wk = (const float*)d_in[3];
    const float* Wv = (const float*)d_in[4];
    const float* Wo = (const float*)d_in[5];
    float* out = (float*)d_out;

    char* ws = (char*)d_ws;
    size_t off = 0;
    auto alloc = [&](size_t bytes) -> void* {
        void* p = ws + off;
        off += (bytes + 255) & ~(size_t)255;
        return p;
    };
    u16* Xb  = (u16*)alloc((size_t)CM * CH * 2);
    u16* Wqb = (u16*)alloc((size_t)CH * CQN * 2);
    u16* Wkb = (u16*)alloc((size_t)CH * CKVN * 2);
    u16* Wvb = (u16*)alloc((size_t)CH * CKVN * 2);
    u16* Wob = (u16*)alloc((size_t)CQN * CH * 2);
    u16* Qb  = (u16*)alloc((size_t)CM * CQN * 2);
    u16* Kc  = (u16*)alloc((size_t)CM * CKVN * 2);
    u16* Vc  = (u16*)alloc((size_t)CM * CKVN * 2);
    u16* Ab  = (u16*)alloc((size_t)CM * CQN * 2);

    auto cvt = [&](const float* s, u16* d, int n) {
        cvt_f32_bf16_kernel<<<(n + 255) / 256, 256, 0, stream>>>(s, d, n);
    };
    cvt(hs, Xb, CM * CH);
    cvt(Wq, Wqb, CH * CQN);
    cvt(Wk, Wkb, CH * CKVN);
    cvt(Wv, Wvb, CH * CKVN);
    cvt(Wo, Wob, CQN * CH);

    gemm_bf16_wmma_kernel<false><<<dim3(CM / 64, CQN / 64), 256, 0, stream>>>(
        Xb, Wqb, nullptr, Qb, CM, CQN, CH);
    gemm_bf16_wmma_kernel<false><<<dim3(CM / 64, CKVN / 64), 256, 0, stream>>>(
        Xb, Wkb, nullptr, Kc, CM, CKVN, CH);
    gemm_bf16_wmma_kernel<false><<<dim3(CM / 64, CKVN / 64), 256, 0, stream>>>(
        Xb, Wvb, nullptr, Vc, CM, CKVN, CH);

    {
        int nq = CM * CNH * 64;
        rope_bf16_kernel<<<(nq + 255) / 256, 256, 0, stream>>>(Qb, CM, CNH);
        int nk = CM * CNKV * 64;
        rope_bf16_kernel<<<(nk + 255) / 256, 256, 0, stream>>>(Kc, CM, CNKV);
    }

    // Flash attention: 4096 waves, 2 waves/block (64 threads)
    flash_attn_wmma_kernel<<<(CB * CNH * (CS / 16)) / 2, 64, 0, stream>>>(Qb, Kc, Vc, Ab);

    gemm_bf16_wmma_kernel<true><<<dim3(CM / 64, CH / 64), 256, 0, stream>>>(
        Ab, Wob, out, nullptr, CM, CH, CQN);
}
